// ExtendedGATLayer_67070209294881
// MI455X (gfx1250) — matrix operand order repaired and checked by reading, not compiled, not run on hardware
//
#include <hip/hip_runtime.h>

#define NN      50000
#define IN_F    128
#define OUT_F   256
#define HEADS   8
#define EE      800000
#define TOTE    (EE + NN)       // 850000 edges incl. self loops
#define HCOLS   768             // [xl | xr | xres]
#define CT      48              // 768/16 column tiles
#define KS      4               // 128/32 k-steps

typedef __attribute__((ext_vector_type(16))) __bf16 v16bf;
typedef __attribute__((ext_vector_type(8)))  float  v8f;

union ABFrag { uint4 q[2]; v16bf v; };

__device__ __forceinline__ unsigned short f2bf(float f) {
    unsigned u = __float_as_uint(f);
    u += 0x7FFFu + ((u >> 16) & 1u);          // round-to-nearest-even
    return (unsigned short)(u >> 16);
}

// ---------------- prep: x fp32 -> bf16 ----------------
__global__ __launch_bounds__(256) void k_convert_x(const float* __restrict__ x,
                                                   unsigned short* __restrict__ xb) {
    int i = blockIdx.x * 256 + threadIdx.x;             // group of 4 elems
    if (i >= (NN * IN_F) / 4) return;
    float4 f = ((const float4*)x)[i];
    ushort4 o;
    o.x = f2bf(f.x); o.y = f2bf(f.y); o.z = f2bf(f.z); o.w = f2bf(f.w);
    ((ushort4*)xb)[i] = o;
}

// ---------------- prep: pack [Wl|Wr|Wres] into B-fragment order ----------------
// wpk[((t*4+s)*32 + lane)*16 + e] = W[k][col], col = t*16+(lane&15),
// k = s*32 + (lane>=16 ? 16 : 0) + e   (contiguous k order, same as A in k_gemm)
__global__ __launch_bounds__(256) void k_pack_w(const float* __restrict__ Wl,
                                                const float* __restrict__ Wr,
                                                const float* __restrict__ Wres,
                                                unsigned short* __restrict__ wpk) {
    int tid = blockIdx.x * 256 + threadIdx.x;
    if (tid >= CT * KS * 32 * 16) return;
    int e    = tid & 15;
    int lane = (tid >> 4) & 31;
    int s    = (tid >> 9) & 3;
    int t    = tid >> 11;
    int col  = t * 16 + (lane & 15);
    int k    = s * 32 + ((lane & 16) ? 16 : 0) + e;
    float w;
    if (col < 256)      w = Wl[k * 256 + col];
    else if (col < 512) w = Wr[k * 256 + (col - 256)];
    else                w = Wres[k * 256 + (col - 512)];
    wpk[tid] = f2bf(w);
}

// ---------------- init: zero accumulator, -inf segment max ----------------
__global__ __launch_bounds__(256) void k_init(float* __restrict__ out,
                                              unsigned* __restrict__ lmax,
                                              float* __restrict__ denom) {
    int i = blockIdx.x * 256 + threadIdx.x;
    if (i < NN * OUT_F) out[i] = 0.f;
    if (i < NN * HEADS) { lmax[i] = 0x007FFFFFu; denom[i] = 0.f; }  // enc(-inf)
}

// ---------------- fused projection GEMM: hcat = x @ [Wl|Wr|Wres] (+bl,+br) ----------------
// grid (3125, 6), 256 threads = 8 waves; each wave: one 16x16 tile, K=128 in 4 WMMAs.
__global__ __launch_bounds__(256) void k_gemm(const unsigned short* __restrict__ xb,
                                              const unsigned short* __restrict__ wpk,
                                              const float* __restrict__ bl,
                                              const float* __restrict__ br,
                                              float* __restrict__ hcat) {
    int lane    = threadIdx.x & 31;
    int wid     = threadIdx.x >> 5;
    int rowTile = blockIdx.x;
    int colBase = blockIdx.y * 128 + wid * 16;
    int t       = colBase >> 4;

    int rowA = rowTile * 16 + (lane & 15);
    int kbA  = (lane & 16) ? 16 : 0;         // A layout: lanes>=16 start at K+16

    v8f c = {};
#pragma unroll
    for (int s = 0; s < KS; ++s) {
        ABFrag a, b;
        const unsigned short* pa = xb + (size_t)rowA * IN_F + s * 32 + kbA;
        a.q[0] = *(const uint4*)(pa);          // K = kb .. kb+7
        a.q[1] = *(const uint4*)(pa + 8);      // K = kb+8 .. kb+15
        const unsigned short* pb = wpk + ((size_t)(t * KS + s) * 32 + lane) * 16;
        b.q[0] = *(const uint4*)(pb);
        b.q[1] = *(const uint4*)(pb + 8);
        c = __builtin_amdgcn_wmma_f32_16x16x32_bf16(false, a.v, false, b.v,
                                                    (short)0, c, false, false);
    }

    int col  = colBase + (lane & 15);
    int row0 = rowTile * 16 + ((lane & 16) ? 8 : 0);
    float bv = 0.f;
    if (col < 256)      bv = bl[col];
    else if (col < 512) bv = br[col - 256];
#pragma unroll
    for (int r = 0; r < 8; ++r)
        hcat[(size_t)(row0 + r) * HCOLS + col] = c[r] + bv;
}

// ---------------- edge pass A: logits + segment max ----------------
__global__ __launch_bounds__(256) void k_edge_logits(const int* __restrict__ ei,
                                                     const float* __restrict__ hcat,
                                                     const float* __restrict__ att,
                                                     float* __restrict__ logits,
                                                     unsigned* __restrict__ lmax) {
    int lane = threadIdx.x & 31;
    long long edge = (long long)blockIdx.x * 8 + (threadIdx.x >> 5);
    if (edge >= TOTE) return;
    int src, dst;
    if (edge < EE) { src = ei[edge]; dst = ei[EE + edge]; }
    else           { src = dst = (int)(edge - EE); }

    int c0 = lane * 8;                                   // 8 channels/lane, 1 head / 4 lanes
    const float* pl = hcat + (size_t)src * HCOLS + c0;         // xl
    const float* pr = hcat + (size_t)dst * HCOLS + 256 + c0;   // xr
    float4 a0 = *(const float4*)pl,       a1 = *(const float4*)(pl + 4);
    float4 b0 = *(const float4*)pr,       b1 = *(const float4*)(pr + 4);
    float4 t0 = *(const float4*)(att + c0), t1 = *(const float4*)(att + c0 + 4);

    float xs[8] = {a0.x + b0.x, a0.y + b0.y, a0.z + b0.z, a0.w + b0.w,
                   a1.x + b1.x, a1.y + b1.y, a1.z + b1.z, a1.w + b1.w};
    float at[8] = {t0.x, t0.y, t0.z, t0.w, t1.x, t1.y, t1.z, t1.w};
    float p = 0.f;
#pragma unroll
    for (int j = 0; j < 8; ++j) {
        float v  = xs[j];
        float lr = v > 0.f ? v : 0.2f * v;               // leaky_relu(0.2)
        p += at[j] * lr;
    }
    p += __shfl_xor(p, 1, 32);
    p += __shfl_xor(p, 2, 32);
    if ((lane & 3) == 0) {
        int h = lane >> 2;
        logits[edge * HEADS + h] = p;
        unsigned bpat = __float_as_uint(p);
        unsigned enc  = bpat ^ ((bpat & 0x80000000u) ? 0xFFFFFFFFu : 0x80000000u);
        atomicMax(&lmax[(size_t)dst * HEADS + h], enc);
    }
}

// ---------------- edge pass B: unnormalized softmax aggregate ----------------
__global__ __launch_bounds__(256) void k_edge_aggregate(const int* __restrict__ ei,
                                                        const float* __restrict__ hcat,
                                                        const float* __restrict__ logits,
                                                        const unsigned* __restrict__ lmax,
                                                        float* __restrict__ denom,
                                                        float* __restrict__ out) {
    int lane = threadIdx.x & 31;
    long long edge = (long long)blockIdx.x * 8 + (threadIdx.x >> 5);
    if (edge >= TOTE) return;
    int src, dst;
    if (edge < EE) { src = ei[edge]; dst = ei[EE + edge]; }
    else           { src = dst = (int)(edge - EE); }

    int h = lane >> 2;
    float lg = logits[edge * HEADS + h];
    unsigned enc  = lmax[(size_t)dst * HEADS + h];
    unsigned bpat = enc ^ ((enc & 0x80000000u) ? 0x80000000u : 0xFFFFFFFFu);
    float ex = __expf(lg - __uint_as_float(bpat));
    if ((lane & 3) == 0) atomicAdd(&denom[(size_t)dst * HEADS + h], ex);

    int c0 = lane * 8;
    const float* pl = hcat + (size_t)src * HCOLS + c0;
    float4 a0 = *(const float4*)pl, a1 = *(const float4*)(pl + 4);
    float* po = out + (size_t)dst * OUT_F + c0;
    atomicAdd(po + 0, ex * a0.x); atomicAdd(po + 1, ex * a0.y);
    atomicAdd(po + 2, ex * a0.z); atomicAdd(po + 3, ex * a0.w);
    atomicAdd(po + 4, ex * a1.x); atomicAdd(po + 5, ex * a1.y);
    atomicAdd(po + 6, ex * a1.z); atomicAdd(po + 7, ex * a1.w);
}

// ---------------- finalize: /denom + residual + bias + LayerNorm + GELU(tanh) ----------------
__global__ __launch_bounds__(256) void k_finalize(float* __restrict__ out,
                                                  const float* __restrict__ hcat,
                                                  const float* __restrict__ denom,
                                                  const float* __restrict__ bias,
                                                  const float* __restrict__ gamma,
                                                  const float* __restrict__ beta) {
    int n = blockIdx.x, c = threadIdx.x;
    int lane = c & 31, wid = c >> 5;
    float agg = out[(size_t)n * OUT_F + c];
    float den = denom[(size_t)n * HEADS + (c >> 5)];
    float v = agg / den + hcat[(size_t)n * HCOLS + 512 + c] + bias[c];

    __shared__ float red[8];
    float s = v;
#pragma unroll
    for (int o = 16; o; o >>= 1) s += __shfl_xor(s, o, 32);
    if (lane == 0) red[wid] = s;
    __syncthreads();
    float tot = 0.f;
#pragma unroll
    for (int i = 0; i < 8; ++i) tot += red[i];
    float mu = tot * (1.f / OUT_F);
    __syncthreads();

    float d = v - mu;
    s = d * d;
#pragma unroll
    for (int o = 16; o; o >>= 1) s += __shfl_xor(s, o, 32);
    if (lane == 0) red[wid] = s;
    __syncthreads();
    tot = 0.f;
#pragma unroll
    for (int i = 0; i < 8; ++i) tot += red[i];
    float var = tot * (1.f / OUT_F);

    float hn = d * rsqrtf(var + 1e-5f) * gamma[c] + beta[c];
    float g  = 0.5f * hn * (1.f + tanhf(0.7978845608f * (hn + 0.044715f * hn * hn * hn)));
    out[(size_t)n * OUT_F + c] = g;
}

extern "C" void kernel_launch(void* const* d_in, const int* in_sizes, int n_in,
                              void* d_out, int out_size, void* d_ws, size_t ws_size,
                              hipStream_t stream) {
    const float* x    = (const float*)d_in[0];
    const int*   ei   = (const int*)d_in[1];
    const float* Wl   = (const float*)d_in[2];
    const float* bl   = (const float*)d_in[3];
    const float* Wr   = (const float*)d_in[4];
    const float* br   = (const float*)d_in[5];
    const float* att  = (const float*)d_in[6];
    const float* Wres = (const float*)d_in[7];
    const float* bias = (const float*)d_in[8];
    const float* lng  = (const float*)d_in[9];
    const float* lnb  = (const float*)d_in[10];
    float* out = (float*)d_out;

    char* ws = (char*)d_ws;
    size_t off = 0;
    auto alloc = [&](size_t bytes) -> void* {
        void* p = ws + off;
        off = (off + bytes + 255) & ~(size_t)255;
        return p;
    };
    unsigned short* xb     = (unsigned short*)alloc((size_t)NN * IN_F * 2);
    unsigned short* wpk    = (unsigned short*)alloc((size_t)CT * KS * 32 * 16 * 2);
    float*          hcat   = (float*)alloc((size_t)NN * HCOLS * 4);
    float*          logits = (float*)alloc((size_t)TOTE * HEADS * 4);
    unsigned*       lmax   = (unsigned*)alloc((size_t)NN * HEADS * 4);
    float*          denom  = (float*)alloc((size_t)NN * HEADS * 4);

    k_convert_x<<<(NN * IN_F / 4 + 255) / 256, 256, 0, stream>>>(x, xb);
    k_pack_w<<<(CT * KS * 32 * 16 + 255) / 256, 256, 0, stream>>>(Wl, Wr, Wres, wpk);
    k_init<<<(NN * OUT_F + 255) / 256, 256, 0, stream>>>(out, lmax, denom);
    k_gemm<<<dim3(NN / 16, HCOLS / 128), 256, 0, stream>>>(xb, wpk, bl, br, hcat);
    k_edge_logits<<<(TOTE + 7) / 8, 256, 0, stream>>>(ei, hcat, att, logits, lmax);
    k_edge_aggregate<<<(TOTE + 7) / 8, 256, 0, stream>>>(ei, hcat, logits, lmax, denom, out);
    k_finalize<<<NN, 256, 0, stream>>>(out, hcat, denom, bias, lng, lnb);
}
